// MaskedMultiHeadLatentRoPESelfAttention_644245095102
// MI455X (gfx1250) — matrix operand order repaired and checked by reading, not compiled, not run on hardware
//
#include <hip/hip_runtime.h>

// ---------------------------------------------------------------------------
// MI455X / gfx1250: wave32, WMMA 16x16x32 bf16 (f32 accumulate),
// double-buffered async global->LDS staging (ASYNCcnt), global prefetch.
// ---------------------------------------------------------------------------
typedef __attribute__((ext_vector_type(16))) __bf16 v16bf;
typedef __attribute__((ext_vector_type(8)))  __bf16 bf16x8;
typedef __attribute__((ext_vector_type(8)))  float  v8f;

#define EMBED 1024
#define PROJ  256
#define QK    64
#define VD    64
#define NH    16
#define BATCH 4
#define SEQ   2048
#define MROWS (BATCH * SEQ)   // 8192

union Frag { v16bf v; bf16x8 h8[2]; };

// async copy of 16 bytes global -> LDS (per lane); tracked with ASYNCcnt
__device__ __forceinline__ void async_copy16(const void* gsrc, void* ldst) {
    const unsigned long long ga = (unsigned long long)(uintptr_t)gsrc;
    // LDS byte address = low 32 bits of the generic pointer (flat truncation)
    const unsigned la = (unsigned)(uintptr_t)ldst;
    asm volatile("global_load_async_to_lds_b128 %0, %1, off"
                 :: "v"(la), "v"(ga) : "memory");
}
__device__ __forceinline__ void async_wait0() {
    asm volatile("s_wait_asynccnt 0x0" ::: "memory");
}

// ---------------------------------------------------------------------------
// f32 -> bf16 pack (n multiple of 4)
// ---------------------------------------------------------------------------
__global__ __launch_bounds__(256)
void pack_bf16(const float* __restrict__ src, __bf16* __restrict__ dst, int n) {
    const int base = (blockIdx.x * 256 + threadIdx.x) * 4;
    if (base < n) {
        float4 v = *(const float4*)(src + base);
        dst[base + 0] = (__bf16)v.x;
        dst[base + 1] = (__bf16)v.y;
        dst[base + 2] = (__bf16)v.z;
        dst[base + 3] = (__bf16)v.w;
    }
}

// f32 [K,N] -> bf16 [N,K] (weights, done once per launch; coalesced reads)
__global__ __launch_bounds__(256)
void transpose_pack_bf16(const float* __restrict__ src, __bf16* __restrict__ dst,
                         int K, int N) {
    const int idx = blockIdx.x * 256 + threadIdx.x;
    if (idx < K * N) {
        const int k = idx / N, n = idx % N;
        dst[(size_t)n * K + k] = (__bf16)src[idx];
    }
}

// ---------------------------------------------------------------------------
// C = A(bf16)[M,K] @ Bt(bf16)[N,K]^T + bias. Optional f32 C and/or bf16 Cbf.
// Block = 4 waves, tile 64x64, K-step 32, double-buffered LDS: async copies
// for tile k+1 overlap the WMMAs of tile k; one barrier per iteration.
// Fragments = two ds_load_b128 each, matching ISA A(16x32)/B(32x16) layouts.
// ---------------------------------------------------------------------------
__global__ __launch_bounds__(128)
void gemm_bf16(const __bf16* __restrict__ A, const __bf16* __restrict__ Bt,
               const float* __restrict__ bias,
               float* __restrict__ C, __bf16* __restrict__ Cbf,
               int M, int K, int N) {
    __shared__ __bf16 As[2][64][32];   // [buf][m][k]
    __shared__ __bf16 Bs[2][64][32];   // [buf][n][k]
    const int t    = threadIdx.x;
    const int w    = t >> 5;
    const int lane = t & 31;
    const int half = lane >> 4;
    const int mr   = lane & 15;
    const int m0   = blockIdx.x * 64;
    const int n0   = blockIdx.y * 64;

    // per-thread staging coordinates: 64 rows x 32 bf16 = 256 x 16B chunks/tile
    const int c0   = t;             // chunk ids t and t+128
    const int row0 = c0 >> 2, off0 = (c0 & 3) * 8;
    const int c1   = t + 128;
    const int row1 = c1 >> 2, off1 = (c1 & 3) * 8;

    auto stage = [&](int k0, int buf) {
        async_copy16(A  + (size_t)(m0 + row0) * K + k0 + off0, &As[buf][row0][off0]);
        async_copy16(Bt + (size_t)(n0 + row0) * K + k0 + off0, &Bs[buf][row0][off0]);
        async_copy16(A  + (size_t)(m0 + row1) * K + k0 + off1, &As[buf][row1][off1]);
        async_copy16(Bt + (size_t)(n0 + row1) * K + k0 + off1, &Bs[buf][row1][off1]);
    };

    v8f acc[4] = {};

    stage(0, 0);
    async_wait0();
    __syncthreads();

    int buf = 0;
    for (int k0 = 0; k0 < K; k0 += 32) {
        // kick off next tile's copies; they run during this tile's WMMAs
        if (k0 + 32 < K) {
            stage(k0 + 32, buf ^ 1);
            if (k0 + 64 < K && t < 64)      // warm L2 two tiles ahead
                __builtin_prefetch(A + (size_t)(m0 + t) * K + k0 + 64, 0, 3);
        }

        Frag af;                 // A 16x32: lane half selects K interleave (0/8)
        const int arow = 16 * w + mr;
        af.h8[0] = *(const bf16x8*)&As[buf][arow][half ? 8 : 0];
        af.h8[1] = *(const bf16x8*)&As[buf][arow][(half ? 8 : 0) + 16];
#pragma unroll
        for (int j = 0; j < 4; ++j) {
            Frag bf;             // B 32x16: lane half selects K interleave (0/8)
            const int brow = 16 * j + mr;
            bf.h8[0] = *(const bf16x8*)&Bs[buf][brow][half ? 8 : 0];
            bf.h8[1] = *(const bf16x8*)&Bs[buf][brow][(half ? 8 : 0) + 16];
            acc[j] = __builtin_amdgcn_wmma_f32_16x16x32_bf16(
                false, af.v, false, bf.v, (short)0, acc[j], false, false);
        }

        async_wait0();     // next tile landed (mostly free: overlapped compute)
        __syncthreads();   // all reads of buf done + all writes to buf^1 visible
        buf ^= 1;
    }

    // epilogue: C layout row = r + 8*half, col = lane%16 within 16x16 tile
#pragma unroll
    for (int j = 0; j < 4; ++j) {
        const int col = n0 + 16 * j + mr;
        const float bb = bias[col];
#pragma unroll
        for (int r = 0; r < 8; ++r) {
            const int row = m0 + 16 * w + r + half * 8;
            const float v = acc[j][r] + bb;
            if (C)   C[(size_t)row * N + col]   = v;
            if (Cbf) Cbf[(size_t)row * N + col] = (__bf16)v;
        }
    }
}

// ---------------------------------------------------------------------------
// RoPE + repack. One wave per (b,s,h): q,k -> head-major bf16 [B,H,S,64],
// v -> dim-major bf16 [B,H,64,S] so attention V B-fragments are contiguous.
// ---------------------------------------------------------------------------
__global__ __launch_bounds__(256)
void rope_pack(const float* __restrict__ qlin, const float* __restrict__ kv,
               __bf16* __restrict__ qbf, __bf16* __restrict__ kbf,
               __bf16* __restrict__ vt) {
    const int gid = blockIdx.x * 256 + threadIdx.x;
    const int wid = gid >> 5;           // (b*S + s)*H + h
    const int j   = gid & 31;           // rotary pair index 0..31
    const int h   = wid & (NH - 1);
    const int bs  = wid >> 4;           // b*S + s
    const int s   = bs & (SEQ - 1);
    const int b   = bs >> 11;

    const float ang = (float)s * __expf(-(float)j * 0.28782313662425576f); // ln(1e4)/32
    float sn, cs;
    __sincosf(ang, &sn, &cs);

    const size_t qoff = (size_t)bs * (QK * NH) + h * QK;
    const float q0 = qlin[qoff + j];
    const float q1 = qlin[qoff + j + 32];
    const size_t koff = (size_t)bs * ((QK + VD) * NH) + h * QK;
    const float k0 = kv[koff + j];
    const float k1 = kv[koff + j + 32];
    const size_t voff = (size_t)bs * ((QK + VD) * NH) + QK * NH + h * VD;
    const float v0 = kv[voff + j];
    const float v1 = kv[voff + j + 32];

    const size_t hm = (((size_t)b * NH + h) * SEQ + s) * QK;
    qbf[hm + j]      = (__bf16)(q0 * cs - q1 * sn);
    qbf[hm + j + 32] = (__bf16)(q1 * cs + q0 * sn);
    kbf[hm + j]      = (__bf16)(k0 * cs - k1 * sn);
    kbf[hm + j + 32] = (__bf16)(k1 * cs + k0 * sn);

    const size_t vtb = ((size_t)b * NH + h) * VD;
    vt[(vtb + j)      * SEQ + s] = (__bf16)v0;
    vt[(vtb + j + 32) * SEQ + s] = (__bf16)v1;
}

// ---------------------------------------------------------------------------
// Causal flash attention. Block = 4 waves = 64 q rows. Per 32-key step:
// 4 WMMA (Q*K^T), online softmax f32, P via per-wave LDS reshape, 4 WMMA (P*V).
// Output written directly in bf16 for the output projection GEMM.
// ---------------------------------------------------------------------------
__global__ __launch_bounds__(128)
void flash_attn(const __bf16* __restrict__ Q, const __bf16* __restrict__ Kb,
                const __bf16* __restrict__ Vt, __bf16* __restrict__ O) {
    __shared__ __bf16 P[4][16][32];
    const int t    = threadIdx.x;
    const int w    = t >> 5;
    const int lane = t & 31;
    const int half = lane >> 4;
    const int mr   = lane & 15;

    const int tile = blockIdx.x & 31;   // S/64 q tiles
    const int bh   = blockIdx.x >> 5;   // b*NH + h
    const int b    = bh >> 4;
    const int h    = bh & 15;
    const int q0   = tile * 64;
    const int qrow = q0 + 16 * w;

    Frag qa[2];
#pragma unroll
    for (int kf = 0; kf < 2; ++kf) {
        const __bf16* p = Q + ((size_t)bh * SEQ + qrow + mr) * QK
                            + 32 * kf + (half ? 8 : 0);
        qa[kf].h8[0] = *(const bf16x8*)p;
        qa[kf].h8[1] = *(const bf16x8*)(p + 16);
    }

    v8f o[4] = {};
    float m[8], l[8];
#pragma unroll
    for (int r = 0; r < 8; ++r) { m[r] = -__builtin_inff(); l[r] = 0.0f; }

    const float scale = 0.125f;         // 1/sqrt(64)
    const int kend = q0 + 64;

    for (int ks = 0; ks < kend; ks += 32) {
        v8f sc[2];
#pragma unroll
        for (int j2 = 0; j2 < 2; ++j2) {
            v8f c = {};
#pragma unroll
            for (int kf = 0; kf < 2; ++kf) {
                Frag kb;   // B 32x16: col = key ks+16*j2+mr, K = head dim
                const __bf16* p = Kb + ((size_t)bh * SEQ + ks + 16 * j2 + mr) * QK
                                     + 32 * kf + (half ? 8 : 0);
                kb.h8[0] = *(const bf16x8*)p;
                kb.h8[1] = *(const bf16x8*)(p + 16);
                c = __builtin_amdgcn_wmma_f32_16x16x32_bf16(
                    false, qa[kf].v, false, kb.v, (short)0, c, false, false);
            }
            sc[j2] = c;
        }

#pragma unroll
        for (int r = 0; r < 8; ++r) {
            const int grow = qrow + r + half * 8;
            float s0 = sc[0][r] * scale;
            float s1 = sc[1][r] * scale;
            if (ks + mr > grow)      s0 = -__builtin_inff();
            if (ks + 16 + mr > grow) s1 = -__builtin_inff();
            float mx = fmaxf(s0, s1);
            mx = fmaxf(mx, __shfl_xor(mx, 1, 16));
            mx = fmaxf(mx, __shfl_xor(mx, 2, 16));
            mx = fmaxf(mx, __shfl_xor(mx, 4, 16));
            mx = fmaxf(mx, __shfl_xor(mx, 8, 16));
            const float mnew = fmaxf(m[r], mx);
            const float corr = __expf(m[r] - mnew);
            const float p0 = __expf(s0 - mnew);
            const float p1 = __expf(s1 - mnew);
            float rs = p0 + p1;
            rs += __shfl_xor(rs, 1, 16);
            rs += __shfl_xor(rs, 2, 16);
            rs += __shfl_xor(rs, 4, 16);
            rs += __shfl_xor(rs, 8, 16);
            l[r] = l[r] * corr + rs;
            m[r] = mnew;
#pragma unroll
            for (int j = 0; j < 4; ++j) o[j][r] *= corr;
            P[w][r + half * 8][mr]      = (__bf16)p0;
            P[w][r + half * 8][16 + mr] = (__bf16)p1;
        }
        __syncthreads();

        Frag pa;   // P as A 16x32
        {
            const __bf16* p = &P[w][mr][half ? 8 : 0];
            pa.h8[0] = *(const bf16x8*)p;
            pa.h8[1] = *(const bf16x8*)(p + 16);
        }
#pragma unroll
        for (int j = 0; j < 4; ++j) {
            Frag vb;   // B[k][n]: k = key offset (dim-major V => contiguous)
            const __bf16* p = Vt + ((size_t)bh * VD + 16 * j + mr) * SEQ
                                 + ks + (half ? 8 : 0);
            vb.h8[0] = *(const bf16x8*)p;
            vb.h8[1] = *(const bf16x8*)(p + 16);
            o[j] = __builtin_amdgcn_wmma_f32_16x16x32_bf16(
                false, pa.v, false, vb.v, (short)0, o[j], false, false);
        }
        __syncthreads();
    }

#pragma unroll
    for (int r = 0; r < 8; ++r) {
        const int grow = qrow + r + half * 8;
        const float inv = 1.0f / l[r];
#pragma unroll
        for (int j = 0; j < 4; ++j) {
            O[((size_t)b * SEQ + grow) * (NH * VD) + h * VD + 16 * j + mr]
                = (__bf16)(o[j][r] * inv);
        }
    }
}

// ---------------------------------------------------------------------------
extern "C" void kernel_launch(void* const* d_in, const int* in_sizes, int n_in,
                              void* d_out, int out_size, void* d_ws, size_t ws_size,
                              hipStream_t stream) {
    (void)in_sizes; (void)n_in; (void)out_size; (void)ws_size;
    const float* x    = (const float*)d_in[0];
    const float* Wq   = (const float*)d_in[1];
    const float* bq   = (const float*)d_in[2];
    const float* Wkv1 = (const float*)d_in[3];
    const float* bkv1 = (const float*)d_in[4];
    const float* Wkv2 = (const float*)d_in[5];
    const float* bkv2 = (const float*)d_in[6];
    const float* Wo   = (const float*)d_in[7];
    const float* bo   = (const float*)d_in[8];
    float* out = (float*)d_out;

    char* ws = (char*)d_ws;
    size_t off = 0;
    auto alloc = [&](size_t bytes) -> void* {
        void* p = ws + off;
        off += (bytes + 255) & ~(size_t)255;
        return p;
    };
    __bf16* xbf    = (__bf16*)alloc((size_t)MROWS * EMBED * 2);
    __bf16* WqT    = (__bf16*)alloc((size_t)EMBED * (QK * NH) * 2);
    __bf16* Wkv1T  = (__bf16*)alloc((size_t)EMBED * PROJ * 2);
    __bf16* Wkv2T  = (__bf16*)alloc((size_t)PROJ * ((QK + VD) * NH) * 2);
    __bf16* WoT    = (__bf16*)alloc((size_t)(NH * VD) * EMBED * 2);
    float*  q_lin  = (float*) alloc((size_t)MROWS * (QK * NH) * 4);
    __bf16* hbf    = (__bf16*)alloc((size_t)MROWS * PROJ * 2);
    float*  kvbuf  = (float*) alloc((size_t)MROWS * ((QK + VD) * NH) * 4);
    __bf16* qbf    = (__bf16*)alloc((size_t)MROWS * (QK * NH) * 2);
    __bf16* kbf    = (__bf16*)alloc((size_t)MROWS * (QK * NH) * 2);
    __bf16* vt     = (__bf16*)alloc((size_t)MROWS * (VD * NH) * 2);
    __bf16* attnbf = (__bf16*)alloc((size_t)MROWS * (VD * NH) * 2);

    const dim3 blk(128);
    // one-time bf16 packing (x) and transposed-weight packing
    pack_bf16<<<(MROWS * EMBED) / 1024, 256, 0, stream>>>(x, xbf, MROWS * EMBED);
    transpose_pack_bf16<<<(EMBED * QK * NH) / 256, 256, 0, stream>>>(Wq, WqT, EMBED, QK * NH);
    transpose_pack_bf16<<<(EMBED * PROJ) / 256, 256, 0, stream>>>(Wkv1, Wkv1T, EMBED, PROJ);
    transpose_pack_bf16<<<(PROJ * (QK + VD) * NH) / 256, 256, 0, stream>>>(Wkv2, Wkv2T, PROJ, (QK + VD) * NH);
    transpose_pack_bf16<<<(NH * VD * EMBED) / 256, 256, 0, stream>>>(Wo, WoT, NH * VD, EMBED);

    // q = x @ Wq + bq            (f32 out for RoPE)
    gemm_bf16<<<dim3(MROWS / 64, (QK * NH) / 64), blk, 0, stream>>>(
        xbf, WqT, bq, q_lin, nullptr, MROWS, EMBED, QK * NH);
    // h = x @ Wkv1 + bkv1        (bf16 out, feeds next GEMM)
    gemm_bf16<<<dim3(MROWS / 64, PROJ / 64), blk, 0, stream>>>(
        xbf, Wkv1T, bkv1, nullptr, hbf, MROWS, EMBED, PROJ);
    // kv = h @ Wkv2 + bkv2       (f32 out for RoPE/V)
    gemm_bf16<<<dim3(MROWS / 64, ((QK + VD) * NH) / 64), blk, 0, stream>>>(
        hbf, Wkv2T, bkv2, kvbuf, nullptr, MROWS, PROJ, (QK + VD) * NH);
    // RoPE + head-major / dim-major repack
    rope_pack<<<(MROWS * NH * 32) / 256, 256, 0, stream>>>(q_lin, kvbuf, qbf, kbf, vt);
    // causal flash attention (bf16 out)
    flash_attn<<<BATCH * NH * (SEQ / 64), blk, 0, stream>>>(qbf, kbf, vt, attnbf);
    // out = attn @ Wo + bo
    gemm_bf16<<<dim3(MROWS / 64, EMBED / 64), blk, 0, stream>>>(
        attnbf, WoT, bo, out, nullptr, MROWS, EMBED, EMBED);
}
